// _NonLocalBlockND_37014028157103
// MI455X (gfx1250) — compile-verified, matrix operand order checked by reading, not run on hardware
//
#include <hip/hip_runtime.h>

// ---------- types ----------
typedef __attribute__((ext_vector_type(16))) __bf16 v16bf;
typedef __attribute__((ext_vector_type(8)))  float  v8f;

#define B_  8
#define C_  256
#define CI_ 128
#define NQ_ 4096   // 64*64
#define NK_ 1024   // 32*32

// ---------- helpers ----------
__device__ inline unsigned short f2bf(float f) {
    unsigned u = __builtin_bit_cast(unsigned, f);
    u += 0x7FFFu + ((u >> 16) & 1u);           // round-to-nearest-even
    return (unsigned short)(u >> 16);
}
__device__ inline float bf2f(unsigned short h) {
    return __builtin_bit_cast(float, ((unsigned)h) << 16);
}
// Load one 16x32 bf16 A-frag (or its mirrored B-frag): two contiguous 16B chunks.
// p points at this lane's first 8 bf16 (K = koff..koff+7); second chunk at +16 elems.
__device__ inline v16bf load_frag(const unsigned short* p) {
    union { uint4 u[2]; v16bf v; } t;
    t.u[0] = *(const uint4*)(p);
    t.u[1] = *(const uint4*)(p + 16);
    return t.v;
}
__device__ inline v8f wmma_bf16(v16bf a, v16bf b, v8f c) {
    return __builtin_amdgcn_wmma_f32_16x16x32_bf16(false, a, false, b, (short)0, c, false, false);
}
__device__ inline float rowmax16(float v) {
    v = fmaxf(v, __shfl_xor(v, 1, 32));
    v = fmaxf(v, __shfl_xor(v, 2, 32));
    v = fmaxf(v, __shfl_xor(v, 4, 32));
    v = fmaxf(v, __shfl_xor(v, 8, 32));
    return v;
}
__device__ inline float rowsum16(float v) {
    v += __shfl_xor(v, 1, 32);
    v += __shfl_xor(v, 2, 32);
    v += __shfl_xor(v, 4, 32);
    v += __shfl_xor(v, 8, 32);
    return v;
}

// ---------- K0: weight conversion + BN folding ----------
__global__ void prep_kernel(const float* g_w, const float* theta_w, const float* phi_w,
                            const float* w_w, const float* w_b,
                            const float* gamma, const float* beta,
                            const float* mean, const float* var,
                            unsigned short* gW, unsigned short* thW,
                            unsigned short* phW, unsigned short* wW,
                            float* alpha, float* beta2) {
    int t = blockIdx.x * 256 + threadIdx.x;           // 32768 threads
    gW[t]  = f2bf(g_w[t]);
    thW[t] = f2bf(theta_w[t]);
    phW[t] = f2bf(phi_w[t]);
    wW[t]  = f2bf(w_w[t]);
    if (t < C_) {
        float s = gamma[t] * rsqrtf(var[t] + 1e-5f);
        alpha[t] = s;
        beta2[t] = beta[t] - mean[t] * s + w_b[t] * s;
    }
}

// ---------- K1/K2: conv1x1 (M=16 pixels, N=128 ci, K=256 c), pixel-major bf16 out ----------
// DUAL: two weight sets sharing one input tile (theta+phi from neighborFeat).
template <bool DUAL>
__global__ __launch_bounds__(128, 1)
void conv1x1_kernel(const float* __restrict__ src,                       // (B,256,4096) f32
                    const unsigned short* __restrict__ w0, const float* __restrict__ b0,
                    unsigned short* __restrict__ out0,                   // (B,4096,128) bf16
                    const unsigned short* __restrict__ w1, const float* __restrict__ b1,
                    unsigned short* __restrict__ out1) {
    __shared__ unsigned short tile[4][16 * C_];       // per-wave 16 pix x 256 c bf16 (8KB)
    const int wid = threadIdx.x >> 5, lane = threadIdx.x & 31;
    const int gwave = blockIdx.x * 4 + wid;           // 2048 wave-tiles
    const int b  = gwave >> 8;
    const int p0 = (gwave & 255) * 16;
    unsigned short* lds = tile[wid];

    // --- load input tile, transpose to (pixel, c) bf16 in LDS ---
    const float* sbase = src + (size_t)b * C_ * NQ_ + p0;
    const int pq = (lane & 3) * 4;                    // pixel quad base
    const int cb = lane >> 2;                         // 0..7
    #pragma unroll 4
    for (int it = 0; it < 32; ++it) {
        const int c = it * 8 + cb;
        const float4 v = *(const float4*)(sbase + (size_t)c * NQ_ + pq);
        lds[(pq + 0) * C_ + c] = f2bf(v.x);
        lds[(pq + 1) * C_ + c] = f2bf(v.y);
        lds[(pq + 2) * C_ + c] = f2bf(v.z);
        lds[(pq + 3) * C_ + c] = f2bf(v.w);
    }
    // wave-private LDS region: LDS ops are in-order per wave, no barrier needed.

    const int row   = lane & 15;
    const int koff  = (lane >> 4) * 8;
    const int rbase = (lane >> 4) * 8;
    const int col   = lane & 15;

    v8f acc0[8] = {}, acc1[8] = {};
    #pragma unroll
    for (int kk = 0; kk < 8; ++kk) {                  // K = 256 in 8 steps of 32
        const v16bf a = load_frag(lds + row * C_ + kk * 32 + koff);
        #pragma unroll
        for (int nb = 0; nb < 8; ++nb) {
            const size_t wo = (size_t)(nb * 16 + row) * C_ + kk * 32 + koff;
            const v16bf bw0 = load_frag(w0 + wo);
            if (DUAL) {
                const v16bf bw1 = load_frag(w1 + wo);
                acc0[nb] = wmma_bf16(a, bw0, acc0[nb]);
                acc1[nb] = wmma_bf16(a, bw1, acc1[nb]);
            } else {
                acc0[nb] = wmma_bf16(a, bw0, acc0[nb]);
            }
        }
    }
    #pragma unroll
    for (int nb = 0; nb < 8; ++nb) {
        const int   ci  = nb * 16 + col;
        const float bb0 = b0[ci];
        const float bb1 = DUAL ? b1[ci] : 0.f;
        #pragma unroll
        for (int j = 0; j < 8; ++j) {
            const size_t o = ((size_t)b * NQ_ + p0 + rbase + j) * CI_ + ci;
            out0[o] = f2bf(acc0[nb][j] + bb0);
            if (DUAL) out1[o] = f2bf(acc1[nb][j] + bb1);
        }
    }
}

// ---------- K3: 2x2 maxpool; phi -> key-major (B,1024,128), g -> ci-major (B,128,1024) ----------
__global__ void pool_kernel(const unsigned short* __restrict__ convPhiT,
                            const unsigned short* __restrict__ convGT,
                            unsigned short* __restrict__ phiT,
                            unsigned short* __restrict__ gMat) {
    const int t  = blockIdx.x * 256 + threadIdx.x;    // 8*1024*128
    const int ci = t & 127;
    const int p  = (t >> 7) & 1023;
    const int b  = t >> 17;
    const int ph = p >> 5, pw = p & 31;
    const int n00 = (ph * 2) * 64 + pw * 2;
    const size_t base = ((size_t)b * NQ_) * CI_ + ci;
    float a0 = bf2f(convPhiT[base + (size_t)(n00)      * CI_]);
    float a1 = bf2f(convPhiT[base + (size_t)(n00 + 1)  * CI_]);
    float a2 = bf2f(convPhiT[base + (size_t)(n00 + 64) * CI_]);
    float a3 = bf2f(convPhiT[base + (size_t)(n00 + 65) * CI_]);
    float g0 = bf2f(convGT [base + (size_t)(n00)      * CI_]);
    float g1 = bf2f(convGT [base + (size_t)(n00 + 1)  * CI_]);
    float g2 = bf2f(convGT [base + (size_t)(n00 + 64) * CI_]);
    float g3 = bf2f(convGT [base + (size_t)(n00 + 65) * CI_]);
    phiT[((size_t)b * NK_ + p) * CI_ + ci] = f2bf(fmaxf(fmaxf(a0, a1), fmaxf(a2, a3)));
    gMat[((size_t)b * CI_ + ci) * NK_ + p] = f2bf(fmaxf(fmaxf(g0, g1), fmaxf(g2, g3)));
}

// ---------- K4: flash attention (16 queries/wave, Nkv=1024 in chunks of 32) ----------
__global__ __launch_bounds__(128, 1)
void attn_kernel(const unsigned short* __restrict__ thetaT,  // (B,4096,128)
                 const unsigned short* __restrict__ phiT,    // (B,1024,128)
                 const unsigned short* __restrict__ gMat,    // (B,128,1024)
                 unsigned short* __restrict__ yT) {          // (B,4096,128)
    __shared__ unsigned short ldsP[4][16 * 32];       // per-wave P tile (1KB)
    const int wid = threadIdx.x >> 5, lane = threadIdx.x & 31;
    unsigned short* lp = ldsP[wid];
    const int gwave = blockIdx.x * 4 + wid;           // 2048 wave-tiles
    const int b  = gwave >> 8;
    const int q0 = (gwave & 255) * 16;
    const int row   = lane & 15;
    const int koff  = (lane >> 4) * 8;
    const int rbase = (lane >> 4) * 8;
    const int col   = lane & 15;

    // Q frags: 16x128 bf16 = 4 A-frags of K=32
    v16bf aT[4];
    const unsigned short* pth = thetaT + ((size_t)b * NQ_ + q0 + row) * CI_;
    #pragma unroll
    for (int kk = 0; kk < 4; ++kk) aT[kk] = load_frag(pth + kk * 32 + koff);

    const unsigned short* phiBase = phiT + ((size_t)b * NK_ + row) * CI_ + koff;
    const unsigned short* gBase   = gMat + ((size_t)b * CI_ + row) * NK_ + koff;

    v8f m, l = {}, acc[8] = {};
    #pragma unroll
    for (int j = 0; j < 8; ++j) m[j] = -1e30f;

    for (int kc = 0; kc < 32; ++kc) {
        const int kb = kc * 32;
        // prefetch next chunk of phi (gfx1250 global_prefetch_b8 path)
        if (kc + 1 < 32) __builtin_prefetch(phiBase + (size_t)(kb + 32) * CI_, 0, 3);
        // ---- scores: S = theta @ phi^T over K=128 ----
        v8f s0 = {}, s1 = {};
        #pragma unroll
        for (int kk = 0; kk < 4; ++kk) {
            const unsigned short* pb0 = phiBase + (size_t)kb * CI_ + kk * 32;
            const v16bf bphi0 = load_frag(pb0);
            const v16bf bphi1 = load_frag(pb0 + 16 * CI_);
            s0 = wmma_bf16(aT[kk], bphi0, s0);
            s1 = wmma_bf16(aT[kk], bphi1, s1);
        }
        // ---- online softmax (rows live across 16-lane groups => stats stay aligned) ----
        v8f p0, p1, scale;
        #pragma unroll
        for (int j = 0; j < 8; ++j) {
            const float rm = rowmax16(fmaxf(s0[j], s1[j]));
            const float mn = fmaxf(m[j], rm);
            scale[j] = __expf(m[j] - mn);
            m[j] = mn;
            p0[j] = __expf(s0[j] - mn);
            p1[j] = __expf(s1[j] - mn);
            l[j] = l[j] * scale[j] + rowsum16(p0[j] + p1[j]);
        }
        #pragma unroll
        for (int nb = 0; nb < 8; ++nb)
            #pragma unroll
            for (int j = 0; j < 8; ++j) acc[nb][j] *= scale[j];
        // ---- stage P (C-layout) -> LDS row-major 16x32 bf16, reload as A-frag ----
        #pragma unroll
        for (int j = 0; j < 8; ++j) {
            lp[(rbase + j) * 32 + col]      = f2bf(p0[j]);
            lp[(rbase + j) * 32 + 16 + col] = f2bf(p1[j]);
        }
        const v16bf aP = load_frag(lp + row * 32 + koff);
        // ---- y += P @ g  (K = 32 keys, N = 128 ci in 8 blocks) ----
        #pragma unroll
        for (int nb = 0; nb < 8; ++nb) {
            const v16bf bg = load_frag(gBase + (size_t)(nb * 16) * NK_ + kb);
            acc[nb] = wmma_bf16(aP, bg, acc[nb]);
        }
    }
    #pragma unroll
    for (int nb = 0; nb < 8; ++nb)
        #pragma unroll
        for (int j = 0; j < 8; ++j)
            yT[((size_t)b * NQ_ + q0 + rbase + j) * CI_ + nb * 16 + col] =
                f2bf(acc[nb][j] / l[j]);
}

// ---------- K5: W conv + folded BN + residual; coalesced float4 epilogue ----------
__global__ __launch_bounds__(128, 1)
void wconv_kernel(const unsigned short* __restrict__ yT,   // (B,4096,128) bf16
                  const unsigned short* __restrict__ wW,   // (256,128) bf16
                  const float* __restrict__ alpha, const float* __restrict__ beta2,
                  const float* __restrict__ x,             // (B,256,4096)
                  float* __restrict__ out) {
    const int wid = threadIdx.x >> 5, lane = threadIdx.x & 31;
    const int gwave = blockIdx.x * 4 + wid;
    const int b  = gwave >> 8;
    const int p0 = (gwave & 255) * 16;
    const int row   = lane & 15;
    const int koff  = (lane >> 4) * 8;
    const int rbase = (lane >> 4) * 8;
    const int col   = lane & 15;

    v16bf a[4];
    const unsigned short* pa = yT + ((size_t)b * NQ_ + p0 + row) * CI_;
    #pragma unroll
    for (int kk = 0; kk < 4; ++kk) a[kk] = load_frag(pa + kk * 32 + koff);

    #pragma unroll 4
    for (int nb = 0; nb < 16; ++nb) {                 // N = 256 output channels
        v8f acc = {};
        #pragma unroll
        for (int kk = 0; kk < 4; ++kk) {
            const unsigned short* pw = wW + (size_t)(nb * 16 + row) * CI_ + kk * 32 + koff;
            acc = wmma_bf16(a[kk], load_frag(pw), acc);
        }
        const int   o  = nb * 16 + col;
        const float al = alpha[o], be = beta2[o];
        const size_t obase = ((size_t)b * C_ + o) * NQ_ + p0 + rbase;
        const float4 x0 = *(const float4*)(x + obase);
        const float4 x1 = *(const float4*)(x + obase + 4);
        float4 r0, r1;
        r0.x = acc[0] * al + be + x0.x;  r0.y = acc[1] * al + be + x0.y;
        r0.z = acc[2] * al + be + x0.z;  r0.w = acc[3] * al + be + x0.w;
        r1.x = acc[4] * al + be + x1.x;  r1.y = acc[5] * al + be + x1.y;
        r1.z = acc[6] * al + be + x1.z;  r1.w = acc[7] * al + be + x1.w;
        *(float4*)(out + obase)     = r0;
        *(float4*)(out + obase + 4) = r1;
    }
}

// ---------- launch ----------
extern "C" void kernel_launch(void* const* d_in, const int* in_sizes, int n_in,
                              void* d_out, int out_size, void* d_ws, size_t ws_size,
                              hipStream_t stream) {
    const float* x      = (const float*)d_in[0];
    const float* nf     = (const float*)d_in[1];
    const float* g_w    = (const float*)d_in[2];
    const float* g_b    = (const float*)d_in[3];
    const float* th_w   = (const float*)d_in[4];
    const float* th_b   = (const float*)d_in[5];
    const float* ph_w   = (const float*)d_in[6];
    const float* ph_b   = (const float*)d_in[7];
    const float* w_w    = (const float*)d_in[8];
    const float* w_b    = (const float*)d_in[9];
    const float* gamma  = (const float*)d_in[10];
    const float* beta   = (const float*)d_in[11];
    const float* mean   = (const float*)d_in[12];
    const float* var    = (const float*)d_in[13];
    float* out = (float*)d_out;

    char* ws = (char*)d_ws;
    auto alloc = [&](size_t bytes) -> char* {
        char* p = ws; ws += (bytes + 255) & ~(size_t)255; return p;
    };
    unsigned short* thW      = (unsigned short*)alloc((size_t)CI_ * C_ * 2);
    unsigned short* phW      = (unsigned short*)alloc((size_t)CI_ * C_ * 2);
    unsigned short* gW       = (unsigned short*)alloc((size_t)CI_ * C_ * 2);
    unsigned short* wW       = (unsigned short*)alloc((size_t)C_ * CI_ * 2);
    float*          alpha    = (float*)alloc(C_ * 4);
    float*          beta2    = (float*)alloc(C_ * 4);
    unsigned short* thetaT   = (unsigned short*)alloc((size_t)B_ * NQ_ * CI_ * 2);
    unsigned short* convPhiT = (unsigned short*)alloc((size_t)B_ * NQ_ * CI_ * 2);
    unsigned short* convGT   = (unsigned short*)alloc((size_t)B_ * NQ_ * CI_ * 2);
    unsigned short* phiT     = (unsigned short*)alloc((size_t)B_ * NK_ * CI_ * 2);
    unsigned short* gMat     = (unsigned short*)alloc((size_t)B_ * CI_ * NK_ * 2);
    unsigned short* yT       = (unsigned short*)alloc((size_t)B_ * NQ_ * CI_ * 2);

    prep_kernel<<<128, 256, 0, stream>>>(g_w, th_w, ph_w, w_w, w_b, gamma, beta, mean, var,
                                         gW, thW, phW, wW, alpha, beta2);
    // theta + phi share the neighborFeat tile
    conv1x1_kernel<true><<<512, 128, 0, stream>>>(nf, thW, th_b, thetaT, phW, ph_b, convPhiT);
    // g from x
    conv1x1_kernel<false><<<512, 128, 0, stream>>>(x, gW, g_b, convGT, nullptr, nullptr, nullptr);
    pool_kernel<<<4096, 256, 0, stream>>>(convPhiT, convGT, phiT, gMat);
    attn_kernel<<<512, 128, 0, stream>>>(thetaT, phiT, gMat, yT);
    wconv_kernel<<<512, 128, 0, stream>>>(yT, wW, alpha, beta2, x, out);
}
